// Memory_41790031790266
// MI455X (gfx1250) — matrix operand order repaired and checked by reading, not compiled, not run on hardware
//
#include <hip/hip_runtime.h>

// Shapes: B=8, HW=M=4096, K=256, C=3. Output (8,4096,260) f32.
// Inputs: d_in[0]=k (8,4096,256), [1]=v (8,4096,3), [2]=rkn_score (8,4096,1),
//         [3]=m_k (8,4096,256), [4]=m_v (8,4096,3), [5]=m_u (8,4096).

typedef __bf16 v16bf __attribute__((ext_vector_type(16)));
typedef float  v8f   __attribute__((ext_vector_type(8)));

#define NB 8
#define NM 4096
#define NK 256
#define NELT (NB * NM * NK)          // 8388608 elements per matrix
#define DECAY_F 0.9f
#define THRESH_F 0.05f

// ---- one-shot f32 -> split-bf16 planes (hi = truncate, lo = RNE residual) ---
__device__ __forceinline__ void split1(float x, unsigned short& h, unsigned short& l) {
  unsigned ub = __float_as_uint(x);
  h = (unsigned short)(ub >> 16);
  float fh = __uint_as_float(((unsigned)h) << 16);
  unsigned ur = __float_as_uint(x - fh);          // residual exact in f32
  l = (unsigned short)((ur + 0x7FFFu + ((ur >> 16) & 1u)) >> 16);
}

__global__ __launch_bounds__(256) void split_bf16_kernel(
    const float* __restrict__ src, unsigned short* __restrict__ hi,
    unsigned short* __restrict__ lo) {
  size_t i = ((size_t)blockIdx.x * 256 + threadIdx.x) * 4;
  float4 f = *(const float4*)(src + i);
  ushort4 h, l;
  split1(f.x, h.x, l.x);
  split1(f.y, h.y, l.y);
  split1(f.z, h.z, l.z);
  split1(f.w, h.w, l.w);
  *(ushort4*)(hi + i) = h;
  *(ushort4*)(lo + i) = l;
}

// ---- WMMA fragment loads from pre-converted planes (no ALU) -----------------
// B fragment (32x16): K {off+0..7, off+16..23} per lane
__device__ __forceinline__ v16bf ld_fragB(const unsigned short* p) {
  union { uint4 q[2]; v16bf v; } u;
  u.q[0] = *(const uint4*)(p);
  u.q[1] = *(const uint4*)(p + 16);
  return u.v;
}
// A fragment (16x32): K {off+0..7, off+16..23} per lane
__device__ __forceinline__ v16bf ld_fragA(const unsigned short* p) {
  union { uint4 q[2]; v16bf v; } u;
  u.q[0] = *(const uint4*)(p);
  u.q[1] = *(const uint4*)(p + 16);
  return u.v;
}

// Two 16x16 f32 tiles (two A row-panels x one B col-panel), K=256 split-bf16.
// 6 WMMA per 32-K chunk against 4 b128 loads; interleaved accumulator chains.
__device__ __forceinline__ void tile_mm2(
    const unsigned short* bh, const unsigned short* bl, int offB,
    const v16bf* a0h, const v16bf* a0l, const v16bf* a1h, const v16bf* a1l,
    v8f& acc0, v8f& acc1) {
  acc0 = {}; acc1 = {};
#pragma unroll
  for (int c = 0; c < 8; ++c) {
    v16bf bhi = ld_fragB(bh + c * 32 + offB);
    v16bf blo = ld_fragB(bl + c * 32 + offB);
    acc0 = __builtin_amdgcn_wmma_f32_16x16x32_bf16(false, a0h[c], false, bhi, (short)0, acc0, false, false);
    acc1 = __builtin_amdgcn_wmma_f32_16x16x32_bf16(false, a1h[c], false, bhi, (short)0, acc1, false, false);
    acc0 = __builtin_amdgcn_wmma_f32_16x16x32_bf16(false, a0h[c], false, blo, (short)0, acc0, false, false);
    acc1 = __builtin_amdgcn_wmma_f32_16x16x32_bf16(false, a1h[c], false, blo, (short)0, acc1, false, false);
    acc0 = __builtin_amdgcn_wmma_f32_16x16x32_bf16(false, a0l[c], false, bhi, (short)0, acc0, false, false);
    acc1 = __builtin_amdgcn_wmma_f32_16x16x32_bf16(false, a1l[c], false, bhi, (short)0, acc1, false, false);
  }
}

// ---- GEMM passes -----------------------------------------------------------
// grid = 8 batches * 16 row-blocks; block = 256 threads (8 waves * 32 rows)
// PASS 1: online softmax row stats -> lse[b,h], max_s_hw[b,h] = 1/Z
// PASS 2: colmax_bits[b,m] = bits( exp( max_h (l[h,m] - lse[h]) ) ) via atomics
template <int PASS>
__global__ __launch_bounds__(256) void gemm_pass_kernel(
    const unsigned short* __restrict__ KH, const unsigned short* __restrict__ KL,
    const unsigned short* __restrict__ MH, const unsigned short* __restrict__ ML,
    float* __restrict__ lse, float* __restrict__ msh,
    unsigned int* __restrict__ colmax) {
  const int b    = blockIdx.x >> 4;
  const int rb   = (blockIdx.x & 15) * 256;
  const int wave = threadIdx.x >> 5;
  const int lane = threadIdx.x & 31;
  const int half = (lane >> 4) & 1;
  const int r0   = rb + wave * 32;            // this wave: rows r0 .. r0+31
  const int offA = half * 8;
  const int offB = half * 8;

  const size_t abase0 = ((size_t)b * NM + r0 + (lane & 15)) * NK;
  const size_t abase1 = abase0 + (size_t)16 * NK;
  v16bf a0h[8], a0l[8], a1h[8], a1l[8];
#pragma unroll
  for (int c = 0; c < 8; ++c) {
    a0h[c] = ld_fragA(KH + abase0 + c * 32 + offA);
    a0l[c] = ld_fragA(KL + abase0 + c * 32 + offA);
    a1h[c] = ld_fragA(KH + abase1 + c * 32 + offA);
    a1l[c] = ld_fragA(KL + abase1 + c * 32 + offA);
  }

  if constexpr (PASS == 1) {
    float rmax[16], rsum[16];
#pragma unroll
    for (int j = 0; j < 16; ++j) { rmax[j] = -3.4e38f; rsum[j] = 0.f; }
    for (int nt = 0; nt < 256; ++nt) {
      const size_t bbase = ((size_t)b * NM + nt * 16 + (lane & 15)) * NK;
      v8f acc0, acc1;
      tile_mm2(MH + bbase, ML + bbase, offB, a0h, a0l, a1h, a1l, acc0, acc1);
#pragma unroll
      for (int j = 0; j < 8; ++j) {
        float v0 = acc0[j];
        float m0 = fmaxf(rmax[j], v0);
        rsum[j]  = rsum[j] * __expf(rmax[j] - m0) + __expf(v0 - m0);
        rmax[j]  = m0;
        float v1 = acc1[j];
        float m1 = fmaxf(rmax[j + 8], v1);
        rsum[j + 8] = rsum[j + 8] * __expf(rmax[j + 8] - m1) + __expf(v1 - m1);
        rmax[j + 8] = m1;
      }
    }
    // combine the 16 lanes sharing each row (rows j: lanes 0-15; j+8: 16-31)
#pragma unroll
    for (int s = 0; s < 2; ++s) {
#pragma unroll
      for (int j = 0; j < 8; ++j) {
        float m = rmax[s * 8 + j], su = rsum[s * 8 + j];
#pragma unroll
        for (int d = 1; d < 16; d <<= 1) {
          float mo = __shfl_xor(m, d, 32);
          float so = __shfl_xor(su, d, 32);
          float mn = fmaxf(m, mo);
          su = su * __expf(m - mn) + so * __expf(mo - mn);
          m = mn;
        }
        if ((lane & 15) == 0) {
          int row_ = r0 + s * 16 + half * 8 + j;
          lse[b * NM + row_] = m + __logf(su);  // logsumexp
          msh[b * NM + row_] = 1.0f / su;       // row max of softmax = 1/Z
        }
      }
    }
  } else {
    __shared__ unsigned int s_cm[NM];
    for (int i = threadIdx.x; i < NM; i += 256) s_cm[i] = 0u;
    __syncthreads();
    float lsev[16];
#pragma unroll
    for (int s = 0; s < 2; ++s)
#pragma unroll
      for (int j = 0; j < 8; ++j)
        lsev[s * 8 + j] = lse[b * NM + r0 + s * 16 + half * 8 + j];
    for (int nt = 0; nt < 256; ++nt) {
      const int col = nt * 16 + (lane & 15);
      const size_t bbase = ((size_t)b * NM + col) * NK;
      v8f acc0, acc1;
      tile_mm2(MH + bbase, ML + bbase, offB, a0h, a0l, a1h, a1l, acc0, acc1);
      float t = -3.4e38f;
#pragma unroll
      for (int j = 0; j < 8; ++j) {
        t = fmaxf(t, acc0[j] - lsev[j]);
        t = fmaxf(t, acc1[j] - lsev[j + 8]);
      }
      t = fmaxf(t, __shfl_xor(t, 16, 32));        // both halves -> 32-row max
      if (lane < 16) atomicMax(&s_cm[col], __float_as_uint(__expf(t)));
    }
    __syncthreads();
    for (int i = threadIdx.x; i < NM; i += 256)
      atomicMax(&colmax[b * NM + i], s_cm[i]);  // positive floats: bit order == value order
  }
}

// ---- zero atomic accumulators each launch (graph-replay safe) --------------
__global__ void init_kernel(unsigned int* colmax, int* count) {
  int i = blockIdx.x * 256 + threadIdx.x;
  if (i < NB * NM) colmax[i] = 0u;
  if (i < NB) count[i] = 0;
}

// ---- stable ascending argsort by rank counting ------------------------------
__global__ __launch_bounds__(256) void argsort_rank_kernel(
    const float* __restrict__ keys, int* __restrict__ idx_out) {
  const int b   = blockIdx.x >> 4;
  const int seg = blockIdx.x & 15;
  const float* kb = keys + b * NM;
  __shared__ float sk[NM];
  for (int i = threadIdx.x; i < NM; i += 256) sk[i] = kb[i];
  __syncthreads();
  const int i = seg * 256 + threadIdx.x;
  const float vi = sk[i];
  int rank = 0;
  for (int j = 0; j < NM; ++j) {
    float vj = sk[j];
    rank += (vj < vi) || ((vj == vi) && (j < i));
  }
  idx_out[b * NM + rank] = i;   // ranks are a permutation -> every slot written
}

// ---- wv_bool -> float keys for the pack sort, plus per-batch count ----------
__global__ void prep_kernel(const float* __restrict__ msh,
                            float* __restrict__ notwv, int* __restrict__ count) {
  int idx = blockIdx.x * 256 + threadIdx.x;   // wave never straddles a batch
  int b = idx >> 12;
  bool wv = msh[idx] < THRESH_F;
  notwv[idx] = wv ? 0.0f : 1.0f;
  unsigned long long bal = __ballot(wv);
  if ((threadIdx.x & 31) == 0) atomicAdd(&count[b], (int)__popcll(bal));
}

// ---- final gather / blend ---------------------------------------------------
__global__ __launch_bounds__(256) void assemble_kernel(
    const float* __restrict__ K, const float* __restrict__ V,
    const float* __restrict__ RKN, const float* __restrict__ MK,
    const float* __restrict__ MV, const float* __restrict__ MU,
    const float* __restrict__ colmax_f, const int* __restrict__ idx_u,
    const int* __restrict__ idx_s, const int* __restrict__ pack,
    const int* __restrict__ count, float* __restrict__ out) {
  const int b = blockIdx.x >> 12;
  const int m = blockIdx.x & (NM - 1);
  const int t = threadIdx.x;
  const int cnt  = count[b];
  const bool w   = (m < cnt);
  const int iu   = idx_u[b * NM + m];
  const int ism  = idx_s[b * NM + m];              // rkn_sorted index
  const int pk   = pack[b * NM + m];
  const int srch = idx_s[b * NM + pk];             // k_sorted[pack] = k[idx_s[pack]]
  float* orow = out + ((size_t)b * NM + m) * 260;

  const float* srck = w ? (K  + ((size_t)b * NM + srch) * NK)
                        : (MK + ((size_t)b * NM + iu)   * NK);
  orow[t] = srck[t];
  if (t < 3) {
    const float* srcv = w ? (V  + ((size_t)b * NM + srch) * 3)
                          : (MV + ((size_t)b * NM + iu)   * 3);
    orow[256 + t] = srcv[t];
  }
  if (t == 0) {
    float rkn_s = RKN[b * NM + ism];
    float u = w ? (1.0f + rkn_s)
                : (DECAY_F * MU[b * NM + iu] + colmax_f[b * NM + m] + rkn_s);
    orow[259] = u;
  }
}

extern "C" void kernel_launch(void* const* d_in, const int* in_sizes, int n_in,
                              void* d_out, int out_size, void* d_ws, size_t ws_size,
                              hipStream_t stream) {
  const float* k   = (const float*)d_in[0];
  const float* v   = (const float*)d_in[1];
  const float* rkn = (const float*)d_in[2];
  const float* mk  = (const float*)d_in[3];
  const float* mv  = (const float*)d_in[4];
  const float* mu  = (const float*)d_in[5];
  float* out = (float*)d_out;

  // workspace layout: ~1 MB of stats + 4 x 16 MB bf16 planes (~65 MB total)
  float* ws = (float*)d_ws;
  float*        lse    = ws;                              // (8*4096) f32
  float*        msh    = ws + 32768;                      // max_s_hw
  unsigned int* colmax = (unsigned int*)(ws + 65536);     // max_s_m bits
  int*          idx_u  = (int*)(ws + 98304);
  int*          idx_s  = (int*)(ws + 131072);
  int*          pack   = (int*)(ws + 163840);
  float*        notwv  = ws + 196608;
  int*          count  = (int*)(ws + 229376);
  unsigned short* khi  = (unsigned short*)(ws + 262144);  // 1 MB offset
  unsigned short* klo  = khi + (size_t)NELT;
  unsigned short* mkhi = khi + 2 * (size_t)NELT;
  unsigned short* mklo = khi + 3 * (size_t)NELT;

  init_kernel<<<128, 256, 0, stream>>>(colmax, count);
  split_bf16_kernel<<<NELT / 1024, 256, 0, stream>>>(k, khi, klo);
  split_bf16_kernel<<<NELT / 1024, 256, 0, stream>>>(mk, mkhi, mklo);
  gemm_pass_kernel<1><<<NB * 16, 256, 0, stream>>>(khi, klo, mkhi, mklo, lse, msh, colmax);
  gemm_pass_kernel<2><<<NB * 16, 256, 0, stream>>>(khi, klo, mkhi, mklo, lse, msh, colmax);
  argsort_rank_kernel<<<NB * 16, 256, 0, stream>>>(mu, idx_u);
  argsort_rank_kernel<<<NB * 16, 256, 0, stream>>>(msh, idx_s);
  prep_kernel<<<128, 256, 0, stream>>>(msh, notwv, count);
  argsort_rank_kernel<<<NB * 16, 256, 0, stream>>>(notwv, pack);
  assemble_kernel<<<NB * NM, 256, 0, stream>>>(k, v, rkn, mk, mv, mu,
                                               (const float*)colmax, idx_u, idx_s,
                                               pack, count, out);
}
